// CKAN_54133767799104
// MI455X (gfx1250) — matrix operand order repaired and checked by reading, not compiled, not run on hardware
//
#include <hip/hip_runtime.h>
#include <hip/hip_bf16.h>
#include <math.h>

typedef __attribute__((ext_vector_type(16))) _Float16 v16h;
typedef __attribute__((ext_vector_type(8)))  _Float16 v8h;
typedef __attribute__((ext_vector_type(8)))  float    v8f;

#define B_    4096
#define T_    64
#define DIM_  128
#define NL_   2
#define ATTN_BPB 8   // batch elements per attention block (weights stay resident in LDS)

// ---------------------------------------------------------------------------
// Prep: zero loss accumulators, build f16 N-major (transposed) attention
// weights so WMMA B-fragments are contiguous in LDS.
//   W1t[n][k] = att_W1[k][n]  (k<256, n<128)
//   W2t[n][k] = att_W2[k][n]  (k<128, n<128)
// ---------------------------------------------------------------------------
__global__ void prep_kernel(const float* __restrict__ W1,
                            const float* __restrict__ W2,
                            const float* __restrict__ W3,
                            _Float16* __restrict__ W1t,
                            _Float16* __restrict__ W2t,
                            _Float16* __restrict__ W3h,
                            float* __restrict__ losses) {
    int i = blockIdx.x * blockDim.x + threadIdx.x;
    if (i < 256 * 128) { int n = i >> 8, k = i & 255; W1t[i] = (_Float16)W1[k * 128 + n]; }
    if (i < 128 * 128) { int n = i >> 7, k = i & 127; W2t[i] = (_Float16)W2[k * 128 + n]; }
    if (i < 128) W3h[i] = (_Float16)W3[i];
    if (i < 2) losses[i] = 0.0f;
}

// ---------------------------------------------------------------------------
// ue0_mean[b][d] = mean_t entity_emb[user_h[0][b][t]][d]
// ---------------------------------------------------------------------------
__global__ void mean_kernel(const int* __restrict__ uh0,
                            const float* __restrict__ E,
                            float* __restrict__ ue0_mean) {
    int b = blockIdx.x, d = threadIdx.x;               // blockDim = 128
    const int* hp = uh0 + b * T_;
    float s = 0.0f;
    for (int t = 0; t < T_; ++t) s += E[(long)hp[t] * DIM_ + d];
    ue0_mean[(long)b * DIM_ + d] = s * (1.0f / 64.0f);
}

// ---------------------------------------------------------------------------
// Knowledge attention. One block per (combo, 8 batch elements); the f16
// weights stay resident in LDS across all 8 elements (8x less L2->LDS
// weight traffic than one-element blocks).
//   combo 0/1: user layers 0/1 (output scaled x2: su==1 => ue*su+ue)
//   combo 2/3: item layers 0/1 (sv==1 => ie)
// 256 threads = 8 waves; per element: M=64 (T), N=128, K=256 then K=128.
// ---------------------------------------------------------------------------
__global__ __launch_bounds__(256) void attn_kernel(
    const int* __restrict__ uh, const int* __restrict__ ur, const int* __restrict__ ut,
    const int* __restrict__ ih, const int* __restrict__ ir, const int* __restrict__ it,
    const float* __restrict__ E, const float* __restrict__ R,
    const _Float16* __restrict__ W1t, const _Float16* __restrict__ W2t,
    const _Float16* __restrict__ W3h,
    float* __restrict__ att_out) {
    extern __shared__ char smem_raw[];
    _Float16* Xs  = (_Float16*)smem_raw;       // 64 x 256
    _Float16* W1s = Xs  + 64 * 256;            // 128 x 256 (N-major), resident
    _Float16* W2s = W1s + 128 * 256;           // 128 x 128 (N-major), resident
    _Float16* Y1s = W2s + 128 * 128;           // 64 x 128
    _Float16* Y2s = Y1s + 64 * 128;            // 64 x 128
    float*    wv  = (float*)(Y2s + 64 * 128);  // 64 attention logits (sigmoid)
    float*    av  = wv + 64;                   // 64 softmax weights

    const int combo = blockIdx.y;
    const int layer = combo & 1;
    const int b0 = blockIdx.x * ATTN_BPB;
    const int* hb; const int* rb; const int* tb; float oscale;
    if (combo < 2) { hb = uh; rb = ur; tb = ut; oscale = 2.0f; }
    else           { hb = ih; rb = ir; tb = it; oscale = 1.0f; }
    hb += (long)layer * B_ * T_;
    rb += (long)layer * B_ * T_;
    tb += (long)layer * B_ * T_;

    const int tid = threadIdx.x;

    // ---- stage weights once; resident for all ATTN_BPB batch elements ----
    for (int i = tid; i < 128 * 256; i += 256) W1s[i] = W1t[i];
    for (int i = tid; i < 128 * 128; i += 256) W2s[i] = W2t[i];

    const int wave  = tid >> 5;
    const int lane  = tid & 31;
    const int lrow  = lane & 15;
    const int lhalf = lane >> 4;        // 0 or 1
    const int m0    = (wave & 3) * 16;  // 4 M-tiles
    const int ngrp  = (wave >> 2) * 4;  // waves 0-3: N-tiles 0..3, waves 4-7: 4..7
    const int hi8   = (lane >= 16) ? 8 : 0;

    for (int bi = 0; bi < ATTN_BPB; ++bi) {
        const int b = b0 + bi;
        const int* hp = hb + (long)b * T_;
        const int* rp = rb + (long)b * T_;
        const int* tp = tb + (long)b * T_;

        // ---- stage X = [h | r] as f16 into LDS ----
        for (int i = tid; i < 64 * 128; i += 256) {
            int row = i >> 7, col = i & 127;
            Xs[row * 256 + col]       = (_Float16)E[(long)hp[row] * DIM_ + col];
            Xs[row * 256 + 128 + col] = (_Float16)R[(long)rp[row] * DIM_ + col];
        }
        // ---- prefetch next element's gathered rows behind the GEMMs ----
        if (bi + 1 < ATTN_BPB) {
            const int* hn = hp + T_;   // next b, same layer/side
            const int* rn = rp + T_;
            const int* tn = tp + T_;
            if (tid < 64)                    __builtin_prefetch(&E[(long)hn[tid] * DIM_], 0, 3);
            else if (tid < 128)              __builtin_prefetch(&R[(long)rn[tid - 64] * DIM_], 0, 3);
            else if (tid < 192)              __builtin_prefetch(&E[(long)tn[tid - 128] * DIM_], 0, 3);
        }
        __syncthreads();

        // ---- layer 1: Y1 = relu(X @ W1),  K = 256 (8 k-steps of 32) ----
        {
            v8f acc[4] = {};
            for (int ks = 0; ks < 8; ++ks) {
                const int kb = ks * 32;
                // A fragment (contiguous k order: lanes>=16 carry K+16..K+31)
                const _Float16* p = Xs + (m0 + lrow) * 256 + kb + lhalf * 16;
                v8h lo = *(const v8h*)p;
                v8h hi = *(const v8h*)(p + 8);
                v16h afrag;
                #pragma unroll
                for (int i2 = 0; i2 < 8; ++i2) { afrag[i2] = lo[i2]; afrag[i2 + 8] = hi[i2]; }
                #pragma unroll
                for (int nt = 0; nt < 4; ++nt) {
                    const int n0 = (ngrp + nt) * 16;
                    v16h bfrag = *(const v16h*)(W1s + (n0 + lrow) * 256 + kb + lhalf * 16);
                    acc[nt] = __builtin_amdgcn_wmma_f32_16x16x32_f16(
                        false, afrag, false, bfrag, (short)0, acc[nt], false, false);
                }
            }
            #pragma unroll
            for (int nt = 0; nt < 4; ++nt) {
                const int cc = (ngrp + nt) * 16 + lrow;
                #pragma unroll
                for (int v2 = 0; v2 < 8; ++v2) {
                    float val = acc[nt][v2];
                    Y1s[(m0 + hi8 + v2) * 128 + cc] = (_Float16)(val > 0.0f ? val : 0.0f);
                }
            }
        }
        __syncthreads();

        // ---- layer 2: Y2 = relu(Y1 @ W2),  K = 128 (4 k-steps) ----
        {
            v8f acc[4] = {};
            for (int ks = 0; ks < 4; ++ks) {
                const int kb = ks * 32;
                const _Float16* p = Y1s + (m0 + lrow) * 128 + kb + lhalf * 16;
                v8h lo = *(const v8h*)p;
                v8h hi = *(const v8h*)(p + 8);
                v16h afrag;
                #pragma unroll
                for (int i2 = 0; i2 < 8; ++i2) { afrag[i2] = lo[i2]; afrag[i2 + 8] = hi[i2]; }
                #pragma unroll
                for (int nt = 0; nt < 4; ++nt) {
                    const int n0 = (ngrp + nt) * 16;
                    v16h bfrag = *(const v16h*)(W2s + (n0 + lrow) * 128 + kb + lhalf * 16);
                    acc[nt] = __builtin_amdgcn_wmma_f32_16x16x32_f16(
                        false, afrag, false, bfrag, (short)0, acc[nt], false, false);
                }
            }
            #pragma unroll
            for (int nt = 0; nt < 4; ++nt) {
                const int cc = (ngrp + nt) * 16 + lrow;
                #pragma unroll
                for (int v2 = 0; v2 < 8; ++v2) {
                    float val = acc[nt][v2];
                    Y2s[(m0 + hi8 + v2) * 128 + cc] = (_Float16)(val > 0.0f ? val : 0.0f);
                }
            }
        }
        __syncthreads();

        // ---- layer 3 + sigmoid: w[t] = sigmoid(Y2[t] . W3) ----
        if (tid < 64) {
            float s = 0.0f;
            for (int k = 0; k < 128; ++k) s += (float)Y2s[tid * 128 + k] * (float)W3h[k];
            wv[tid] = 1.0f / (1.0f + expf(-s));
        }
        __syncthreads();

        // ---- softmax over T=64 (redundant per-thread reduction; T is tiny) ----
        if (tid < 64) {
            float mx = -3.0e38f;
            for (int t2 = 0; t2 < 64; ++t2) mx = fmaxf(mx, wv[t2]);
            float den = 0.0f;
            for (int t2 = 0; t2 < 64; ++t2) den += expf(wv[t2] - mx);
            av[tid] = expf(wv[tid] - mx) / den;
        }
        __syncthreads();

        // ---- weighted gather-sum over t, apply (dead-code) layer scale ----
        if (tid < 128) {
            float s = 0.0f;
            for (int t2 = 0; t2 < 64; ++t2) s += av[t2] * E[(long)tp[t2] * DIM_ + tid];
            att_out[((long)combo * B_ + b) * DIM_ + tid] = s * oscale;
        }
        __syncthreads();   // protect Xs/wv/av before next iteration restages
    }
}

// ---------------------------------------------------------------------------
// DAE: noisy = sigmoid(relu((x + 0.3*noise) @ We + be) @ Wd + bd)
// also accumulates mean((x - noisy)^2) into loss_accum.
// x = x_direct[b] (user: ue0_mean) or gather E[gidx[b]] (item).
// ---------------------------------------------------------------------------
__global__ void dae_kernel(const float* __restrict__ x_direct,
                           const int* __restrict__ gidx,
                           const float* __restrict__ E,
                           const float* __restrict__ noise,
                           const float* __restrict__ We, const float* __restrict__ be,
                           const float* __restrict__ Wd, const float* __restrict__ bd,
                           float* __restrict__ noisy_out,
                           float* __restrict__ loss_accum,
                           int use_gather) {
    __shared__ float xs[128], nxs[128], encs[64], red[128];
    int b = blockIdx.x, d = threadIdx.x;               // blockDim = 128
    float xv = use_gather ? E[(long)gidx[b] * DIM_ + d]
                          : x_direct[(long)b * DIM_ + d];
    xs[d]  = xv;
    nxs[d] = xv + 0.3f * noise[(long)b * DIM_ + d];
    __syncthreads();
    if (d < 64) {
        float s = be[d];
        for (int k = 0; k < 128; ++k) s += nxs[k] * We[k * 64 + d];
        encs[d] = fmaxf(s, 0.0f);
    }
    __syncthreads();
    float s = bd[d];
    for (int k = 0; k < 64; ++k) s += encs[k] * Wd[k * 128 + d];
    float nv = 1.0f / (1.0f + expf(-s));
    noisy_out[(long)b * DIM_ + d] = nv;
    float diff = xs[d] - nv;
    red[d] = diff * diff;
    __syncthreads();
    for (int off = 64; off > 0; off >>= 1) {
        if (d < off) red[d] += red[d + off];
        __syncthreads();
    }
    if (d == 0) atomicAdd(loss_accum, red[0] * (1.0f / (4096.0f * 128.0f)));
}

// ---------------------------------------------------------------------------
// predict2: stack reversed embeddings, 3x3 VALID conv (NCHW/OIHW correlation),
// elementwise dot over 126 cols, sigmoid.
// ---------------------------------------------------------------------------
__global__ void predict_kernel(const float* __restrict__ att_out,
                               const float* __restrict__ noisy_user,
                               const int* __restrict__ items,
                               const float* __restrict__ E,
                               const float* __restrict__ cuw, const float* __restrict__ cub,
                               const float* __restrict__ cvw, const float* __restrict__ cvb,
                               float* __restrict__ scores) {
    __shared__ float eu[3][128], ev[3][128], red[128];
    int b = blockIdx.x, d = threadIdx.x;               // blockDim = 128
    // reversed(user_embs) = [2*ue_1, 2*ue_0, noisy_user]  (x2 already folded in)
    eu[0][d] = att_out[((long)1 * B_ + b) * DIM_ + d];
    eu[1][d] = att_out[((long)0 * B_ + b) * DIM_ + d];
    eu[2][d] = noisy_user[(long)b * DIM_ + d];
    // reversed(item_embs) = [ie_1, ie_0, item_emb_origin]
    ev[0][d] = att_out[((long)3 * B_ + b) * DIM_ + d];
    ev[1][d] = att_out[((long)2 * B_ + b) * DIM_ + d];
    ev[2][d] = E[(long)items[b] * DIM_ + d];
    __syncthreads();
    float prod = 0.0f;
    if (d < 126) {
        float u = cub[0], v = cvb[0];
        #pragma unroll
        for (int p = 0; p < 3; ++p)
            #pragma unroll
            for (int q = 0; q < 3; ++q) {
                u += eu[p][d + q] * cuw[p * 3 + q];
                v += ev[p][d + q] * cvw[p * 3 + q];
            }
        prod = u * v;
    }
    red[d] = prod;
    __syncthreads();
    for (int off = 64; off > 0; off >>= 1) {
        if (d < off) red[d] += red[d + off];
        __syncthreads();
    }
    if (d == 0) scores[b] = 1.0f / (1.0f + expf(-red[0]));
}

// ---------------------------------------------------------------------------
extern "C" void kernel_launch(void* const* d_in, const int* in_sizes, int n_in,
                              void* d_out, int out_size, void* d_ws, size_t ws_size,
                              hipStream_t stream) {
    const int*   items      = (const int*)d_in[0];
    const int*   user_h     = (const int*)d_in[1];
    const int*   user_r     = (const int*)d_in[2];
    const int*   user_t     = (const int*)d_in[3];
    const int*   item_h     = (const int*)d_in[4];
    const int*   item_r     = (const int*)d_in[5];
    const int*   item_t     = (const int*)d_in[6];
    const float* user_noise = (const float*)d_in[7];
    const float* item_noise = (const float*)d_in[8];
    const float* E          = (const float*)d_in[9];
    const float* R          = (const float*)d_in[10];
    const float* att_W1     = (const float*)d_in[11];
    const float* att_W2     = (const float*)d_in[12];
    const float* att_W3     = (const float*)d_in[13];
    const float* u_We = (const float*)d_in[14]; const float* u_be = (const float*)d_in[15];
    const float* u_Wd = (const float*)d_in[16]; const float* u_bd = (const float*)d_in[17];
    const float* i_We = (const float*)d_in[18]; const float* i_be = (const float*)d_in[19];
    const float* i_Wd = (const float*)d_in[20]; const float* i_bd = (const float*)d_in[21];
    // d_in[22..25] = WQU/WKU/WQV/WKV: dead code (softmax over size-1 axis == 1)
    const float* convU_w = (const float*)d_in[26]; const float* convU_b = (const float*)d_in[27];
    const float* convV_w = (const float*)d_in[28]; const float* convV_b = (const float*)d_in[29];

    float* out = (float*)d_out;          // [0..4095] scores, [4096] user loss, [4097] item loss

    // ---- workspace carve (~14.1 MB) ----
    float*    att_out    = (float*)d_ws;                        // 4 * B * 128 f32
    float*    ue0_mean   = att_out + 4L * B_ * DIM_;            // B * 128
    float*    noisy_user = ue0_mean + (long)B_ * DIM_;          // B * 128
    float*    noisy_item = noisy_user + (long)B_ * DIM_;        // B * 128
    _Float16* W1t        = (_Float16*)(noisy_item + (long)B_ * DIM_);  // 128*256
    _Float16* W2t        = W1t + 128 * 256;                     // 128*128
    _Float16* W3h        = W2t + 128 * 128;                     // 128

    prep_kernel<<<128, 256, 0, stream>>>(att_W1, att_W2, att_W3, W1t, W2t, W3h, out + B_);

    mean_kernel<<<B_, 128, 0, stream>>>(user_h /* layer 0 */, E, ue0_mean);

    const size_t shmem =
        (size_t)(64 * 256 + 128 * 256 + 128 * 128 + 64 * 128 + 64 * 128) * sizeof(_Float16)
        + 2 * 64 * sizeof(float);
    (void)hipFuncSetAttribute((const void*)attn_kernel,
                              hipFuncAttributeMaxDynamicSharedMemorySize, (int)shmem);
    attn_kernel<<<dim3(B_ / ATTN_BPB, 4), 256, shmem, stream>>>(
        user_h, user_r, user_t, item_h, item_r, item_t, E, R, W1t, W2t, W3h, att_out);

    dae_kernel<<<B_, 128, 0, stream>>>(ue0_mean, nullptr, E, user_noise,
                                       u_We, u_be, u_Wd, u_bd,
                                       noisy_user, out + B_ + 0, 0);
    dae_kernel<<<B_, 128, 0, stream>>>(nullptr, items, E, item_noise,
                                       i_We, i_be, i_Wd, i_bd,
                                       noisy_item, out + B_ + 1, 1);

    predict_kernel<<<B_, 128, 0, stream>>>(att_out, noisy_user, items, E,
                                           convU_w, convU_b, convV_w, convV_b, out);
}
